// MultiHeadAttention_70171175682288
// MI455X (gfx1250) — matrix operand order repaired and checked by reading, not compiled, not run on hardware
//
#include <hip/hip_runtime.h>
#include <cstdint>

// ---------------------------------------------------------------------------
// MHA forward for MI455X (gfx1250): bf16 WMMA flash attention with
// TDM (tensor_load_to_lds) + async global->LDS staging, double buffered.
// ---------------------------------------------------------------------------

typedef __attribute__((ext_vector_type(16))) __bf16 v16bf;
typedef __attribute__((ext_vector_type(8)))  float  v8f;
typedef unsigned int tdm4u __attribute__((ext_vector_type(4)));
typedef int          tdm8i __attribute__((ext_vector_type(8)));
typedef int          tdm4i __attribute__((ext_vector_type(4)));
typedef int          i32x4 __attribute__((ext_vector_type(4)));

constexpr int D_MODEL = 1024;
constexpr int SEQ     = 2048;
constexpr int BATCH   = 2;
constexpr int HEADS   = 16;
constexpr int DK      = 64;
constexpr int MTOT    = BATCH * SEQ;   // 4096
constexpr int KDIM    = D_MODEL;
constexpr int NDIM    = D_MODEL;

// ---- gfx1250 feature probes (host pass sees fallbacks only) ---------------
#if defined(__AMDGCN__) && __has_builtin(__builtin_amdgcn_global_load_async_to_lds_b128) && \
    __has_builtin(__builtin_amdgcn_s_wait_asynccnt)
#define HAVE_ASYNC 1
#else
#define HAVE_ASYNC 0
#endif
#if defined(__AMDGCN__) && __has_builtin(__builtin_amdgcn_tensor_load_to_lds) && \
    __has_builtin(__builtin_amdgcn_s_wait_tensorcnt)
#define HAVE_TDM 1
#else
#define HAVE_TDM 0
#endif
#if HAVE_TDM
#define PER_STAGE 1   // per-thread async ops per stage (V only; K goes via TDM)
#else
#define PER_STAGE 2   // K + V both via per-thread async
#endif

__device__ __forceinline__ __bf16 tobf(float f) { return (__bf16)f; }

__device__ __forceinline__ v8f wmma_bf16(v16bf a, v16bf b, v8f c) {
  return __builtin_amdgcn_wmma_f32_16x16x32_bf16(false, a, false, b, (short)0, c,
                                                 false, false);
}

#if HAVE_ASYNC
// Prototype (from clang diagnostic): (int4 addrspace(1)*, int4 addrspace(3)*, Ii, Ii)
__device__ __forceinline__ void async_b128(__bf16* lds_dst, const __bf16* gsrc) {
  __builtin_amdgcn_global_load_async_to_lds_b128(
      (__attribute__((address_space(1))) i32x4*)(uintptr_t)gsrc,
      (__attribute__((address_space(3))) i32x4*)(unsigned)(uintptr_t)lds_dst,
      0, 0);
}
#endif

#if HAVE_TDM
// TDM: load one contiguous 4 KB K-tile (32 rows x 64 bf16) into LDS.
// D# built per CDNA5 ISA 8.3/8.4: 1-D tile, data_size=2B, tile_dim0=2048.
__device__ __forceinline__ void tdm_load_k(__bf16* lds_dst, const __bf16* gsrc) {
  const unsigned lds = (unsigned)(uintptr_t)lds_dst;
  const unsigned long long ga = (unsigned long long)(uintptr_t)gsrc;
  tdm4u g0;
  g0.x = 1u;                                                  // count=1 (valid)
  g0.y = lds;                                                 // lds_addr
  g0.z = (unsigned)ga;                                        // global_addr lo
  g0.w = (unsigned)((ga >> 32) & 0x1FFFFFFu) | (2u << 30);    // addr hi | type=2
  tdm8i g1;
  g1[0] = (int)(1u << 16);        // data_size = 1 -> 2 bytes
  g1[1] = (int)(2048u << 16);     // tensor_dim0[15:0] = 2048
  g1[2] = (int)(1u << 16);        // tensor_dim0 hi = 0, tensor_dim1 = 1
  g1[3] = (int)(2048u << 16);     // tensor_dim1 hi = 0, tile_dim0 = 2048
  g1[4] = 0;                      // tile_dim1 = 0 (unused), tile_dim2 = 0
  g1[5] = 2048;                   // tensor_dim0_stride lo
  g1[6] = 0;
  g1[7] = 0;
  tdm4i gz = {0, 0, 0, 0};
#if __clang_major__ >= 23
  tdm8i z8 = {0, 0, 0, 0, 0, 0, 0, 0};
  __builtin_amdgcn_tensor_load_to_lds(g0, g1, gz, gz, z8, 0);
#else
  __builtin_amdgcn_tensor_load_to_lds(g0, g1, gz, gz, 0);
#endif
}
#endif

// ---------------------------------------------------------------------------
// GEMM: C[M,N] = A[M,K] @ W[N,K]^T   (M=4096, N=K=1024 for all four uses)
// MODE 0: bf16 row-major (Q,K) | MODE 1: bf16 per-head V^T | MODE 2: f32+bias
// ---------------------------------------------------------------------------
template <int MODE, bool ABF>
__global__ __launch_bounds__(256) void gemm_xwT(const void* __restrict__ Av,
                                                const float* __restrict__ W,
                                                const float* __restrict__ bias,
                                                void* __restrict__ Cout) {
  const int lane = threadIdx.x & 31;
  const int wv   = threadIdx.x >> 5;
  const int g    = lane >> 4;
  const int ln   = lane & 15;
  const int wm   = wv >> 1;
  const int wn   = wv & 1;
  const int m0 = blockIdx.x * 128 + wm * 32;
  const int n0 = blockIdx.y * 64  + wn * 32;

  v8f acc[2][2] = {};

  for (int k0 = 0; k0 < KDIM; k0 += 32) {
    v16bf a[2];
#pragma unroll
    for (int i = 0; i < 2; ++i) {
      const int m = m0 + i * 16 + ln;
      if (ABF) {
        const __bf16* r = (const __bf16*)Av + (size_t)m * KDIM + k0;
        __builtin_prefetch(r + 32, 0, 0);
#pragma unroll
        for (int e = 0; e < 8; ++e) {
          a[i][e]     = r[8 * g + e];
          a[i][8 + e] = r[16 + 8 * g + e];
        }
      } else {
        const float* r = (const float*)Av + (size_t)m * KDIM + k0;
        __builtin_prefetch(r + 32, 0, 0);
#pragma unroll
        for (int e = 0; e < 8; ++e) {
          a[i][e]     = tobf(r[8 * g + e]);
          a[i][8 + e] = tobf(r[16 + 8 * g + e]);
        }
      }
    }
    v16bf bm[2];
#pragma unroll
    for (int j = 0; j < 2; ++j) {
      const int n = n0 + j * 16 + ln;
      const float* r = W + (size_t)n * KDIM + k0;
      __builtin_prefetch(r + 32, 0, 0);
#pragma unroll
      for (int e = 0; e < 8; ++e) { bm[j][e] = tobf(r[8 * g + e]); bm[j][8 + e] = tobf(r[16 + 8 * g + e]); }
    }
#pragma unroll
    for (int i = 0; i < 2; ++i)
#pragma unroll
      for (int j = 0; j < 2; ++j) acc[i][j] = wmma_bf16(a[i], bm[j], acc[i][j]);
  }

#pragma unroll
  for (int i = 0; i < 2; ++i)
#pragma unroll
    for (int j = 0; j < 2; ++j)
#pragma unroll
      for (int r = 0; r < 8; ++r) {
        const int m = m0 + i * 16 + r + 8 * g;
        const int n = n0 + j * 16 + ln;
        const float v = acc[i][j][r];
        if (MODE == 0) {
          ((__bf16*)Cout)[(size_t)m * NDIM + n] = tobf(v);
        } else if (MODE == 1) {
          const int bb  = m >> 11;
          const int rr  = m & (SEQ - 1);
          const int t   = rr * D_MODEL + n;
          const int h   = t >> 17;
          const int rem = t & (SEQ * DK - 1);
          const int s   = rem >> 6;
          const int d   = rem & (DK - 1);
          ((__bf16*)Cout)[(size_t)bb * D_MODEL * SEQ +
                          (size_t)((h << 6) + d) * SEQ + s] = tobf(v);
        } else {
          ((float*)Cout)[(size_t)m * NDIM + n] = v + bias[n];
        }
      }
}

// ---------------------------------------------------------------------------
// Flash attention: block = 8 waves x 16 query rows = 128 rows.
// K/V^T tiles double-buffered in LDS, staged via TDM + async-to-LDS.
// ---------------------------------------------------------------------------
__global__ __launch_bounds__(256) void flash_attn(const void* __restrict__ Qv,
                                                  const void* __restrict__ Kv,
                                                  const void* __restrict__ Vtv,
                                                  void* __restrict__ Hv) {
  const __bf16* Qb = (const __bf16*)Qv;
  const __bf16* Kb = (const __bf16*)Kv;
  const __bf16* Vt = (const __bf16*)Vtv;
  __bf16* Hc = (__bf16*)Hv;

  __shared__ __bf16 Kt[2][32][64];   // key tile, [s][d]
  __shared__ __bf16 Vs[2][64][32];   // V^T tile, [d][s]
  __shared__ __bf16 Pl[8][16][32];   // per-wave P staging

  const int tid  = threadIdx.x;
  const int lane = tid & 31;
  const int w    = tid >> 5;
  const int g    = lane >> 4;
  const int ln   = lane & 15;

  const int bh   = blockIdx.x >> 4;
  const int qblk = blockIdx.x & 15;
  const int b    = bh >> 4;
  const int h    = bh & 15;
  const size_t base  = (size_t)b * SEQ * D_MODEL + (size_t)h * SEQ * DK;
  const size_t vbase = (size_t)b * D_MODEL * SEQ + (size_t)h * DK * SEQ;
  const int q0 = qblk * 128 + w * 16;

  const __bf16* gK = Kb + base;
  const __bf16* gV = Vt + vbase;

  const int krow = tid >> 3, kcol = (tid & 7) * 8;   // 32x64 in 16B chunks
  const int vrow = tid >> 2, vcol = (tid & 3) * 8;   // 64x32 in 16B chunks

  auto stage = [&](int buf, int kt) {
#if HAVE_TDM
    if (w == 0) tdm_load_k(&Kt[buf][0][0], gK + (size_t)kt * DK);
#elif HAVE_ASYNC
    async_b128(&Kt[buf][krow][kcol], gK + (size_t)(kt + krow) * DK + kcol);
#else
    {
      const __bf16* s = gK + (size_t)(kt + krow) * DK + kcol;
      __bf16* d = &Kt[buf][krow][kcol];
#pragma unroll
      for (int e = 0; e < 8; ++e) d[e] = s[e];
    }
#endif
#if HAVE_ASYNC
    async_b128(&Vs[buf][vrow][vcol], gV + (size_t)vrow * SEQ + kt + vcol);
#else
    {
      const __bf16* s = gV + (size_t)vrow * SEQ + kt + vcol;
      __bf16* d = &Vs[buf][vrow][vcol];
#pragma unroll
      for (int e = 0; e < 8; ++e) d[e] = s[e];
    }
#endif
  };

  // Q fragments: 16x64 = two 16x32 A-fragments, loaded once from global.
  v16bf qa[2];
  {
    const __bf16* qr = Qb + base + (size_t)(q0 + ln) * DK;
#pragma unroll
    for (int ks = 0; ks < 2; ++ks)
#pragma unroll
      for (int e = 0; e < 8; ++e) {
        qa[ks][e]     = qr[ks * 32 + 16 * g + e];
        qa[ks][8 + e] = qr[ks * 32 + 16 * g + 8 + e];
      }
  }

  v8f o[4] = {};
  float mst[8], lst[8];
#pragma unroll
  for (int r = 0; r < 8; ++r) { mst[r] = -__builtin_inff(); lst[r] = 0.f; }

  stage(0, 0);  // prologue

  for (int kt = 0; kt < SEQ; kt += 32) {
    const int  buf  = (kt >> 5) & 1;
    const bool more = (kt + 32) < SEQ;
    if (more) stage(buf ^ 1, kt + 32);

#if HAVE_TDM
    if (w == 0) {
      if (more) __builtin_amdgcn_s_wait_tensorcnt(1);
      else      __builtin_amdgcn_s_wait_tensorcnt(0);
    }
#endif
#if HAVE_ASYNC
    if (more) __builtin_amdgcn_s_wait_asynccnt(PER_STAGE);
    else      __builtin_amdgcn_s_wait_asynccnt(0);
#endif
    __syncthreads();   // all threads' tile data for `buf` has landed

    // S-tile 16x32 = Q(16x64) @ K_tile(64x32)^T  (B-frags from LDS)
    v8f sc[2] = {};
#pragma unroll
    for (int j = 0; j < 2; ++j)
#pragma unroll
      for (int ks = 0; ks < 2; ++ks) {
        v16bf bk;
        const __bf16* kr = &Kt[buf][j * 16 + ln][ks * 32 + 16 * g];
#pragma unroll
        for (int e = 0; e < 16; ++e) bk[e] = kr[e];
        sc[j] = wmma_bf16(qa[ks], bk, sc[j]);
      }

    // online softmax (row = 16 lanes; xor<16 stays within half-wave group)
    float p0[8], p1[8], alpha[8];
#pragma unroll
    for (int r = 0; r < 8; ++r) {
      const float s0 = sc[0][r] * 0.125f;
      const float s1 = sc[1][r] * 0.125f;
      float mx = fmaxf(s0, s1);
      mx = fmaxf(mx, __shfl_xor(mx, 1));
      mx = fmaxf(mx, __shfl_xor(mx, 2));
      mx = fmaxf(mx, __shfl_xor(mx, 4));
      mx = fmaxf(mx, __shfl_xor(mx, 8));
      const float mn = fmaxf(mst[r], mx);
      alpha[r] = __expf(mst[r] - mn);
      mst[r]   = mn;
      p0[r] = __expf(s0 - mn);
      p1[r] = __expf(s1 - mn);
      float rs = p0[r] + p1[r];
      rs += __shfl_xor(rs, 1);
      rs += __shfl_xor(rs, 2);
      rs += __shfl_xor(rs, 4);
      rs += __shfl_xor(rs, 8);
      lst[r] = lst[r] * alpha[r] + rs;
    }
#pragma unroll
    for (int nd = 0; nd < 4; ++nd)
#pragma unroll
      for (int r = 0; r < 8; ++r) o[nd][r] *= alpha[r];

    // P: C-layout -> LDS -> A-fragment
#pragma unroll
    for (int r = 0; r < 8; ++r) {
      Pl[w][r + 8 * g][ln]      = tobf(p0[r]);
      Pl[w][r + 8 * g][16 + ln] = tobf(p1[r]);
    }
    v16bf pa;
    {
      const __bf16* pr = &Pl[w][ln][0];
#pragma unroll
      for (int e = 0; e < 8; ++e) {
        pa[e]     = pr[16 * g + e];
        pa[8 + e] = pr[16 * g + 8 + e];
      }
    }

    // O += P(16x32) @ V_tile(32x64), B-frags from LDS V^T rows
#pragma unroll
    for (int nd = 0; nd < 4; ++nd) {
      v16bf bv;
      const __bf16* vr = &Vs[buf][nd * 16 + ln][16 * g];
#pragma unroll
      for (int e = 0; e < 16; ++e) bv[e] = vr[e];
      o[nd] = wmma_bf16(pa, bv, o[nd]);
    }

    __syncthreads();   // everyone done reading `buf` before it is restaged
  }

#pragma unroll
  for (int r = 0; r < 8; ++r) lst[r] = 1.0f / lst[r];
#pragma unroll
  for (int nd = 0; nd < 4; ++nd)
#pragma unroll
    for (int r = 0; r < 8; ++r) {
      const int q = q0 + r + 8 * g;
      Hc[base + (size_t)q * DK + nd * 16 + ln] = tobf(o[nd][r] * lst[r]);
    }
}

// ---------------------------------------------------------------------------
extern "C" void kernel_launch(void* const* d_in, const int* in_sizes, int n_in,
                              void* d_out, int out_size, void* d_ws,
                              size_t ws_size, hipStream_t stream) {
  (void)in_sizes; (void)n_in; (void)out_size; (void)ws_size;
  const float* Xq = (const float*)d_in[0];
  const float* Xk = (const float*)d_in[1];
  const float* Xv = (const float*)d_in[2];
  const float* Wq = (const float*)d_in[3];
  const float* Wk = (const float*)d_in[4];
  const float* Wv = (const float*)d_in[5];
  const float* Wo = (const float*)d_in[6];
  const float* bo = (const float*)d_in[7];

  const size_t NE = (size_t)MTOT * D_MODEL;
  unsigned short* ws = (unsigned short*)d_ws;
  void* Qb = (void*)(ws + 0 * NE);
  void* Kb = (void*)(ws + 1 * NE);
  void* Vt = (void*)(ws + 2 * NE);
  void* Hc = (void*)(ws + 3 * NE);

  dim3 blk(256);
  dim3 grid(MTOT / 128, NDIM / 64);

  gemm_xwT<0, false><<<grid, blk, 0, stream>>>(Xq, Wq, nullptr, Qb);
  gemm_xwT<0, false><<<grid, blk, 0, stream>>>(Xk, Wk, nullptr, Kb);
  gemm_xwT<1, false><<<grid, blk, 0, stream>>>(Xv, Wv, nullptr, Vt);

  flash_attn<<<dim3(BATCH * HEADS * (SEQ / 128)), blk, 0, stream>>>(Qb, Kb, Vt, Hc);

  gemm_xwT<2, true><<<grid, blk, 0, stream>>>(Hc, Wo, bo, d_out);
}
